// CirculantElman_83236466196692
// MI455X (gfx1250) — matrix operand order repaired and checked by reading, not compiled, not run on hardware
//
#include <hip/hip_runtime.h>
#include <hip/hip_bf16.h>
#include <math.h>

// ---------------------------------------------------------------------------
// CirculantElman on MI455X (gfx1250): all dense math through
// v_wmma_f32_16x16x32_bf16 (f32 accumulate). Circulant ops become dense GEMMs
// against materialized 2048x2048 circulant matrices (8MB bf16, L2-resident).
// The T=1024 scan is 1024 graph-captured step launches with ping-pong h.
// GEMM uses double-buffered LDS; global->LDS staging goes through the CDNA5
// async-to-LDS path (ASYNCcnt) when the toolchain exposes the builtins.
// ---------------------------------------------------------------------------

typedef __bf16 bf16_t;
typedef __bf16 bf16x16 __attribute__((ext_vector_type(16)));
typedef float  f32x8   __attribute__((ext_vector_type(8)));
typedef int    v4i_t   __attribute__((ext_vector_type(4)));

union FragAB { bf16x16 v; uint4 q[2]; };

#define DIMX  1024
#define DI    2048
#define BATCH 8
#define TLEN  1024
#define NTOK  (BATCH * TLEN)   // 8192 tokens

// ---- CDNA5 async global->LDS copy (guarded; falls back to sync loads) -----
#if defined(__has_builtin)
#if __has_builtin(__builtin_amdgcn_global_load_async_to_lds_b128) && \
    __has_builtin(__builtin_amdgcn_s_wait_asynccnt)
#define USE_ASYNC_LDS 1
#endif
#endif
#ifndef USE_ASYNC_LDS
#define USE_ASYNC_LDS 0
#endif

#if USE_ASYNC_LDS
// clang prototype: vV4i*1 V4i*3 Ii Ii  (global int4*, LDS int4*, imm, imm)
typedef __attribute__((address_space(1))) v4i_t* as1_v4i_ptr;
typedef __attribute__((address_space(3))) v4i_t* as3_v4i_ptr;

__device__ __forceinline__ void async_copy_b128(const void* g, void* l) {
  // generic->as1: full 64-bit address; generic->as3: low 32 bits = LDS offset
  as1_v4i_ptr gp = (as1_v4i_ptr)(unsigned long long)(uintptr_t)g;
  as3_v4i_ptr lp = (as3_v4i_ptr)(unsigned int)(uintptr_t)l;
  __builtin_amdgcn_global_load_async_to_lds_b128(gp, lp, 0, 0);
}
#endif

// ----------------------------- prep kernels --------------------------------

__global__ void k_f32_to_bf16(const float* __restrict__ src,
                              bf16_t* __restrict__ dst, int n) {
  int i = blockIdx.x * 256 + threadIdx.x;
  if (i < n) dst[i] = (bf16_t)src[i];
}

// W is [N,K] row-major (as in the reference params); Bt[k*N+n] = bf16(W[n*K+k])
__global__ void k_transpose_bf16(const float* __restrict__ W,
                                 bf16_t* __restrict__ Bt, int N, int K) {
  int i = blockIdx.x * 256 + threadIdx.x;
  if (i < N * K) {
    int k = i / N, n = i % N;
    Bt[i] = (bf16_t)W[n * K + k];
  }
}

// C[k*D+n] = bf16(c[(n-k) mod D]); D power of two
__global__ void k_circulant_bf16(const float* __restrict__ c,
                                 bf16_t* __restrict__ C, int D) {
  int i = blockIdx.x * 256 + threadIdx.x;
  if (i < D * D) {
    int k = i / D, n = i % D;
    C[i] = (bf16_t)c[(n - k) & (D - 1)];
  }
}

__global__ void k_init_h(const float* __restrict__ h0, float* hA, float* hB) {
  int i = blockIdx.x * 256 + threadIdx.x;           // 16 x DI (rows 8..15 = pad)
  if (i < 16 * DI) {
    hA[i] = (i < BATCH * DI) ? h0[i] : 0.f;
    hB[i] = 0.f;
  }
}

__global__ void k_copy_h(const float* __restrict__ hsrc, float* __restrict__ dst) {
  int i = blockIdx.x * 256 + threadIdx.x;
  if (i < BATCH * DI) dst[i] = hsrc[i];
}

// ------------------------- main WMMA GEMM kernel ---------------------------
// Out[M,N] = A[M,K](bf16,row-major) x B[K,N](bf16,row-major), f32 accumulate.
// Block: 256 thr = 8 waves; block tile 128x128; wave tile 32(M) x 64(N)
//   -> 2x4 grid of 16x16 WMMA accumulators per wave.
// Double-buffered LDS; async global->LDS when available.
// EPI: 0 = store f32; 1 = store bf16; 2 = f32 silu(acc+bias); 3 = f32 acc+bias
template <int EPI>
__global__ __launch_bounds__(256)
void k_gemm_bf16(const bf16_t* __restrict__ A, const bf16_t* __restrict__ B,
                 const float* __restrict__ bias,
                 float* __restrict__ outF, bf16_t* __restrict__ outB,
                 int M, int N, int K) {
  __shared__ bf16_t sA[2][128 * 32];   // [m][k], row stride 32
  __shared__ bf16_t sB[2][32 * 128];   // [k][n], row stride 128

  const int tid  = threadIdx.x;
  const int lane = tid & 31;
  const int wid  = tid >> 5;
  const int wm   = wid >> 1;                 // 0..3
  const int wn   = wid & 1;                  // 0..1
  const int bm   = blockIdx.y * 128;
  const int bn   = blockIdx.x * 128;

  const f32x8 zero = {0.f, 0.f, 0.f, 0.f, 0.f, 0.f, 0.f, 0.f};
  f32x8 acc[2][4];
#pragma unroll
  for (int i = 0; i < 2; ++i)
#pragma unroll
    for (int j = 0; j < 4; ++j) acc[i][j] = zero;

  const int ksel = (lane >> 4) << 3;         // 0 or 8 (A-fragment K select)
  const int nk   = K >> 5;

  // per-tile staging: 2x uint4 of A + 2x uint4 of B per thread (4 async ops)
  auto issue_tile = [&](int kt, int buf) {
    const int kb = kt << 5;
#pragma unroll
    for (int i = tid; i < 512; i += 256) {
      int r = i >> 2, c = i & 3;             // A: 128 rows x 4 uint4 (32 bf16)
      const uint4* src =
          reinterpret_cast<const uint4*>(A + (size_t)(bm + r) * K + kb) + c;
      uint4* dst = reinterpret_cast<uint4*>(sA[buf] + r * 32) + c;
#if USE_ASYNC_LDS
      async_copy_b128(src, dst);
#else
      *dst = *src;
#endif
    }
#pragma unroll
    for (int i = tid; i < 512; i += 256) {
      int r = i >> 4, c = i & 15;            // B: 32 rows x 16 uint4 (128 bf16)
      const uint4* src =
          reinterpret_cast<const uint4*>(B + (size_t)(kb + r) * N + bn) + c;
      uint4* dst = reinterpret_cast<uint4*>(sB[buf] + r * 128) + c;
#if USE_ASYNC_LDS
      async_copy_b128(src, dst);
#else
      *dst = *src;
#endif
    }
  };

  issue_tile(0, 0);

#pragma unroll 1
  for (int kt = 0; kt < nk; ++kt) {
    const int cur = kt & 1;
    if (kt + 1 < nk) {
      issue_tile(kt + 1, cur ^ 1);           // prefetch next tile into other buf
#if USE_ASYNC_LDS
      __builtin_amdgcn_s_wait_asynccnt(4);   // tile kt's 4 copies have landed
#endif
    } else {
#if USE_ASYNC_LDS
      __builtin_amdgcn_s_wait_asynccnt(0);
#endif
    }
    __syncthreads();                          // all waves' tile-kt data visible

    // ---- fragments per ISA layout ----
    FragAB fa[2], fb[4];
#pragma unroll
    for (int mt = 0; mt < 2; ++mt) {
      int m = wm * 32 + mt * 16 + (lane & 15);
      const uint4* p = reinterpret_cast<const uint4*>(sA[cur] + m * 32 + ksel);
      fa[mt].q[0] = p[0];                    // K = ksel .. ksel+7
      fa[mt].q[1] = p[2];                    // K = ksel+16 .. ksel+23
    }
#pragma unroll
    for (int nt = 0; nt < 4; ++nt) {
      int nb = wn * 64 + nt * 16;
      const bf16_t* p = sB[cur] + (size_t)ksel * 128 + nb + (lane & 15);
#pragma unroll
      for (int s = 0; s < 8; ++s) {          // B col N=lane&15, K=ksel+s / 16+ksel+s
        fb[nt].v[s]     = p[s * 128];
        fb[nt].v[8 + s] = p[(16 + s) * 128];
      }
    }
#pragma unroll
    for (int mt = 0; mt < 2; ++mt)
#pragma unroll
      for (int nt = 0; nt < 4; ++nt)
        acc[mt][nt] = __builtin_amdgcn_wmma_f32_16x16x32_bf16(
            false, fa[mt].v, false, fb[nt].v, (short)0, acc[mt][nt], false, false);
    __syncthreads();                          // done reading buf cur
  }

  // ---- epilogue: D layout -> row = r + 8*(lane>=16), col = lane&15 ----
  const int colLane = lane & 15;
  const int rowSel  = (lane >> 4) << 3;
#pragma unroll
  for (int mt = 0; mt < 2; ++mt) {
#pragma unroll
    for (int nt = 0; nt < 4; ++nt) {
      const int col = bn + wn * 64 + nt * 16 + colLane;
      float bia = 0.f;
      if (EPI == 2 || EPI == 3) bia = bias[col];
#pragma unroll
      for (int r = 0; r < 8; ++r) {
        const int row = bm + wm * 32 + mt * 16 + rowSel + r;
        const size_t o = (size_t)row * N + col;
        const float v = acc[mt][nt][r];
        if (EPI == 0) {
          outF[o] = v;
        } else if (EPI == 1) {
          outB[o] = (bf16_t)v;
        } else if (EPI == 2) {
          const float z = v + bia;
          outF[o] = z / (1.f + __expf(-z));  // silu
        } else {
          outF[o] = v + bia;
        }
      }
    }
  }
}

// ------------------------------ scan step ----------------------------------
// One recurrence step: H_new[16,2048] = tanh(H_prev @ C_h + pre_t), rows 0..7
// valid (batch). Block: 256 thr = 8 waves, each wave one 16-col tile; grid 16.
__global__ __launch_bounds__(256)
void k_step(const float* __restrict__ hprev, float* __restrict__ hnext,
            const float* __restrict__ pre, const float* __restrict__ gate,
            const bf16_t* __restrict__ Ch, bf16_t* __restrict__ cell, int t) {
  __shared__ bf16_t sH[16 * DI];             // 64KB: whole h_prev, bf16
  const int tid = threadIdx.x;
  for (int i = tid; i < 16 * DI; i += 256) sH[i] = (bf16_t)hprev[i];
  __syncthreads();

  const int lane    = tid & 31;
  const int wid     = tid >> 5;
  const int colBase = blockIdx.x * 128 + wid * 16;
  const int ksel    = (lane >> 4) << 3;
  const int m       = lane & 15;

  f32x8 acc = {0.f, 0.f, 0.f, 0.f, 0.f, 0.f, 0.f, 0.f};
#pragma unroll 4
  for (int kt = 0; kt < DI / 32; ++kt) {
    const int kb = kt * 32;
    // hide L2 latency on next K-tile's circulant rows (global_prefetch_b8)
    if (kt + 1 < DI / 32)
      __builtin_prefetch(Ch + (size_t)(kb + 32 + lane) * DI + colBase, 0, 1);
    FragAB fa, fb;
    const uint4* pa = reinterpret_cast<const uint4*>(sH + m * DI + kb + ksel);
    fa.q[0] = pa[0];
    fa.q[1] = pa[2];
    const bf16_t* pb = Ch + (size_t)(kb + ksel) * DI + colBase + m;
#pragma unroll
    for (int s = 0; s < 8; ++s) {
      fb.v[s]     = pb[(size_t)s * DI];
      fb.v[8 + s] = pb[(size_t)(16 + s) * DI];
    }
    acc = __builtin_amdgcn_wmma_f32_16x16x32_bf16(false, fa.v, false, fb.v,
                                                  (short)0, acc, false, false);
  }

  const int rowSel = (lane >> 4) << 3;
  const int col    = colBase + (lane & 15);
#pragma unroll
  for (int r = 0; r < 8; ++r) {
    const int row = rowSel + r;              // 0..15; only batch rows written
    if (row < BATCH) {
      const size_t gr = (size_t)(row * TLEN + t) * DI + col;
      const float v = tanhf(acc[r] + pre[gr]);
      hnext[row * DI + col] = v;
      cell[gr] = (bf16_t)(v * gate[gr]);
    }
  }
}

// ------------------------------- launcher ----------------------------------

extern "C" void kernel_launch(void* const* d_in, const int* in_sizes, int n_in,
                              void* d_out, int out_size, void* d_ws,
                              size_t ws_size, hipStream_t stream) {
  (void)in_sizes; (void)n_in; (void)out_size; (void)ws_size;
  const float* x    = (const float*)d_in[0];
  const float* h0   = (const float*)d_in[1];
  const float* Win  = (const float*)d_in[2];   // [DI, DIMX]
  const float* Wout = (const float*)d_in[3];   // [DIMX, DI]
  const float* ch   = (const float*)d_in[4];
  const float* cx   = (const float*)d_in[5];
  const float* b    = (const float*)d_in[6];
  const float* Wg   = (const float*)d_in[7];   // [DI, DI]
  const float* bg   = (const float*)d_in[8];

  float* out   = (float*)d_out;                         // [B,T,DIMX]
  float* out_h = out + (size_t)BATCH * TLEN * DIMX;     // [B,DI]

  char* w = (char*)d_ws;
  auto carve = [&](size_t bytes) -> char* {
    char* p = w;
    w += (bytes + 255) & ~(size_t)255;
    return p;
  };
  bf16_t* x_b   = (bf16_t*)carve(sizeof(bf16_t) * (size_t)NTOK * DIMX);
  bf16_t* xp_b  = (bf16_t*)carve(sizeof(bf16_t) * (size_t)NTOK * DI);
  bf16_t* WinT  = (bf16_t*)carve(sizeof(bf16_t) * (size_t)DIMX * DI);
  bf16_t* WoutT = (bf16_t*)carve(sizeof(bf16_t) * (size_t)DI * DIMX);
  bf16_t* WgT   = (bf16_t*)carve(sizeof(bf16_t) * (size_t)DI * DI);
  bf16_t* Cx    = (bf16_t*)carve(sizeof(bf16_t) * (size_t)DI * DI);
  bf16_t* Ch    = (bf16_t*)carve(sizeof(bf16_t) * (size_t)DI * DI);
  float*  gate  = (float*)carve(sizeof(float) * (size_t)NTOK * DI);
  float*  pre   = (float*)carve(sizeof(float) * (size_t)NTOK * DI);
  bf16_t* cell  = (bf16_t*)carve(sizeof(bf16_t) * (size_t)NTOK * DI);
  float*  hA    = (float*)carve(sizeof(float) * 16 * DI);
  float*  hB    = (float*)carve(sizeof(float) * 16 * DI);

  const dim3 blk(256);
  // prep
  k_f32_to_bf16<<<(NTOK * DIMX + 255) / 256, blk, 0, stream>>>(x, x_b, NTOK * DIMX);
  k_transpose_bf16<<<(DI * DIMX + 255) / 256, blk, 0, stream>>>(Win, WinT, DI, DIMX);
  k_transpose_bf16<<<(DI * DIMX + 255) / 256, blk, 0, stream>>>(Wout, WoutT, DIMX, DI);
  k_transpose_bf16<<<(DI * DI + 255) / 256, blk, 0, stream>>>(Wg, WgT, DI, DI);
  k_circulant_bf16<<<(DI * DI + 255) / 256, blk, 0, stream>>>(cx, Cx, DI);
  k_circulant_bf16<<<(DI * DI + 255) / 256, blk, 0, stream>>>(ch, Ch, DI);

  // x_proj (bf16 out), gate (silu), pre (+b)
  const dim3 g1(DI / 128, NTOK / 128);                  // (16, 64)
  k_gemm_bf16<1><<<g1, blk, 0, stream>>>(x_b, WinT, nullptr, nullptr, xp_b,
                                         NTOK, DI, DIMX);
  k_gemm_bf16<2><<<g1, blk, 0, stream>>>(xp_b, WgT, bg, gate, nullptr,
                                         NTOK, DI, DI);
  k_gemm_bf16<3><<<g1, blk, 0, stream>>>(xp_b, Cx, b, pre, nullptr,
                                         NTOK, DI, DI);

  // sequential scan, ping-pong h
  k_init_h<<<(16 * DI + 255) / 256, blk, 0, stream>>>(h0, hA, hB);
  for (int t = 0; t < TLEN; ++t) {
    const float* hin = (t & 1) ? hB : hA;
    float* hout      = (t & 1) ? hA : hB;
    k_step<<<DI / 128, blk, 0, stream>>>(hin, hout, pre, gate, Ch, cell, t);
  }

  // output projection + final h (t=1023 wrote into hA)
  const dim3 g2(DIMX / 128, NTOK / 128);                // (8, 64)
  k_gemm_bf16<0><<<g2, blk, 0, stream>>>(cell, WoutT, nullptr, out, nullptr,
                                         NTOK, DIMX, DI);
  k_copy_h<<<(BATCH * DI + 255) / 256, blk, 0, stream>>>(hA, out_h);
}
